// Encoder_9234179687250
// MI455X (gfx1250) — matrix operand order repaired and checked by reading, not compiled, not run on hardware
//
#include <hip/hip_runtime.h>
#include <hip/hip_bf16.h>
#include <math.h>
#include <stdint.h>

// ---------------------------------------------------------------- types
typedef __bf16 bf16;
typedef bf16  v16bf __attribute__((ext_vector_type(16)));
typedef bf16  v8bf  __attribute__((ext_vector_type(8)));
typedef float v8f   __attribute__((ext_vector_type(8)));
typedef int   v4i   __attribute__((ext_vector_type(4)));

// ---------------------------------------------------------------- sizes
static constexpr int BB   = 64;
static constexpr int SS   = 512;
static constexpr int TOK  = BB * SS;   // 32768
static constexpr int KPAD = 864;       // 850 padded to 27*32
static constexpr int HID  = 512;
static constexpr int G4   = 2048;      // 4 * RNN_H
static constexpr int RNNH = 512;

// ---------------------------------------------------------------- async LDS path (guarded)
#if __has_builtin(__builtin_amdgcn_global_load_async_to_lds_b128) && \
    __has_builtin(__builtin_amdgcn_s_wait_asynccnt)
#define HAVE_ASYNC_LDS 1
#else
#define HAVE_ASYNC_LDS 0
#endif

#if HAVE_ASYNC_LDS
typedef __attribute__((address_space(1))) v4i* gv4i_p;   // global int4*
typedef __attribute__((address_space(3))) v4i* lv4i_p;   // LDS int4*
#endif

__device__ __forceinline__ void copy16_g2lds(const bf16* __restrict__ g, bf16* l) {
#if HAVE_ASYNC_LDS
  __builtin_amdgcn_global_load_async_to_lds_b128((gv4i_p)g, (lv4i_p)l, 0, 0);
#else
  *(v8bf*)l = *(const v8bf*)g;
#endif
}
__device__ __forceinline__ void wait_lds_copies() {
#if HAVE_ASYNC_LDS
  __builtin_amdgcn_s_wait_asynccnt(0);
#endif
}

// ---------------------------------------------------------------- WMMA helpers
__device__ __forceinline__ v8f wmma_bf16(v16bf a, v16bf b, v8f c) {
  return __builtin_amdgcn_wmma_f32_16x16x32_bf16(false, a, false, b, (short)0, c,
                                                 false, false);
}

// A-matrix fragment (16x32 bf16): lane<16 -> row=lane, k = kb+0..15
//                                 lane>=16 -> row=lane-16, k = kb+16..31
__device__ __forceinline__ v16bf load_a_frag(const bf16* __restrict__ base,
                                             int stride, int m0, int kb, int lane) {
  int row = m0 + (lane & 15);
  int kc  = kb + ((lane >> 4) << 4);
  const bf16* p = base + (size_t)row * stride + kc;
  union { v16bf v; v8bf h[2]; } u;
  u.h[0] = *(const v8bf*)(p);
  u.h[1] = *(const v8bf*)(p + 8);
  return u.v;
}

// B-matrix fragment (32x16 bf16): lane<16 -> col=lane, k = kb+0..15
//                                 lane>=16 -> col=lane-16, k = kb+16..31
__device__ __forceinline__ v16bf load_b_frag(const bf16* __restrict__ w,
                                             int stride, int n0, int kb, int lane) {
  int n  = n0 + (lane & 15);
  int kc = kb + ((lane >> 4) << 4);
  return *(const v16bf*)(w + (size_t)n * stride + kc);
}

__device__ __forceinline__ float sigf(float x) { return 1.0f / (1.0f + __expf(-x)); }

// ---------------------------------------------------------------- small kernels
__global__ void zero32_kernel(uint32_t* p, int n) {
  int i = blockIdx.x * blockDim.x + threadIdx.x;
  if (i < n) p[i] = 0u;
}

__global__ void cvt_kernel(const float* __restrict__ src, bf16* __restrict__ dst, int n) {
  int i = blockIdx.x * blockDim.x + threadIdx.x;
  if (i < n) dst[i] = (bf16)src[i];
}

// W_lin [512,850] f32 -> [512,864] bf16 (zero pad)
__global__ void cvt_pad_wlin_kernel(const float* __restrict__ src, bf16* __restrict__ dst) {
  int i = blockIdx.x * blockDim.x + threadIdx.x;
  if (i < HID * KPAD) {
    int r = i / KPAD, c = i % KPAD;
    dst[i] = (c < 850) ? (bf16)src[r * 850 + c] : (bf16)0.0f;
  }
}

// ---------------------------------------------------------------- gather+concat
// catL/catR [32768,864] bf16, row m = b*512+s.
// order: char(200) extChar(200) {l|r}Bi(200) {l|r}ExtBi(200) ctype(50) pad(14)
__global__ void gather_kernel(const int* __restrict__ char_idx,
                              const int* __restrict__ ctype_idx,
                              const int* __restrict__ lbi_idx,
                              const int* __restrict__ rbi_idx,
                              const int* __restrict__ ext_idx,
                              const int* __restrict__ lebi_idx,
                              const int* __restrict__ rebi_idx,
                              const float* __restrict__ charEmb,
                              const float* __restrict__ ctypeEmb,
                              const float* __restrict__ biEmb,
                              const float* __restrict__ extEmb,
                              const float* __restrict__ extBiEmb,
                              bf16* __restrict__ catL, bf16* __restrict__ catR) {
  int tok = blockIdx.x;
  int ci = char_idx[tok], ct = ctype_idx[tok];
  int lb = lbi_idx[tok],  rb = rbi_idx[tok];
  int ei = ext_idx[tok];
  int le = lebi_idx[tok], re = rebi_idx[tok];
  bf16* rowL = catL + (size_t)tok * KPAD;
  bf16* rowR = catR + (size_t)tok * KPAD;
  for (int c = threadIdx.x; c < KPAD; c += blockDim.x) {
    float vl, vr;
    if (c < 200)      { vl = vr = charEmb[(size_t)ci * 200 + c]; }
    else if (c < 400) { vl = vr = extEmb[(size_t)ei * 200 + (c - 200)]; }
    else if (c < 600) { vl = biEmb[(size_t)lb * 200 + (c - 400)];
                        vr = biEmb[(size_t)rb * 200 + (c - 400)]; }
    else if (c < 800) { vl = extBiEmb[(size_t)le * 200 + (c - 600)];
                        vr = extBiEmb[(size_t)re * 200 + (c - 600)]; }
    else if (c < 850) { vl = vr = ctypeEmb[(size_t)ct * 50 + (c - 800)]; }
    else              { vl = vr = 0.0f; }
    rowL[c] = (bf16)vl;
    rowR[c] = (bf16)vr;
  }
}

// ---------------------------------------------------------------- GEMM1
// in = tanh(cat @ W_lin^T + b_lin)   M=32768 N=512 K=864
// Ping-pong fragment buffers indexed by (ks & 1); unroll-2 makes the index a
// compile-time constant in each loop copy -> no register copies, loads for
// iteration k+1 overlap the WMMAs of iteration k.
__global__ __launch_bounds__(256) void gemm_lin_kernel(
    const bf16* __restrict__ catL, const bf16* __restrict__ catR,
    const bf16* __restrict__ wlin, const float* __restrict__ b_lin,
    bf16* __restrict__ outL, bf16* __restrict__ outR) {
  const bf16* cat = blockIdx.z ? catR : catL;
  bf16*       out = blockIdx.z ? outR : outL;
  int wave = threadIdx.x >> 5, lane = threadIdx.x & 31;
  int m0 = blockIdx.y * 128 + wave * 16;
  int n0 = blockIdx.x * 64;
  constexpr int KS = KPAD / 32;
  v8f acc[4] = {};
  v16bf a[2], b[2][4];
  a[0] = load_a_frag(cat, KPAD, m0, 0, lane);
#pragma unroll
  for (int nt = 0; nt < 4; ++nt) b[0][nt] = load_b_frag(wlin, KPAD, n0 + nt * 16, 0, lane);
#pragma unroll 2
  for (int ks = 0; ks < KS; ++ks) {
    const int cur = ks & 1, nxt = cur ^ 1;
    if (ks + 1 < KS) {
      int kb = (ks + 1) * 32;
      __builtin_prefetch(cat + (size_t)(m0 + (lane & 15)) * KPAD + kb + 32, 0, 1);
      a[nxt] = load_a_frag(cat, KPAD, m0, kb, lane);
#pragma unroll
      for (int nt = 0; nt < 4; ++nt)
        b[nxt][nt] = load_b_frag(wlin, KPAD, n0 + nt * 16, kb, lane);
    }
#pragma unroll
    for (int nt = 0; nt < 4; ++nt) acc[nt] = wmma_bf16(a[cur], b[cur][nt], acc[nt]);
  }
#pragma unroll
  for (int nt = 0; nt < 4; ++nt) {
    int col = n0 + nt * 16 + (lane & 15);
    float bias = b_lin[col];
#pragma unroll
    for (int r = 0; r < 8; ++r) {
      int row = m0 + r + ((lane >> 4) << 3);
      out[(size_t)row * HID + col] = (bf16)tanhf(acc[nt][r] + bias);
    }
  }
}

// ---------------------------------------------------------------- GEMM2
// xg[s][b][n] = in[b*S+s] @ W_ih^T + b_ih + b_hh   M=32768 N=2048 K=512
__global__ __launch_bounds__(256) void gemm_ih_kernel(
    const bf16* __restrict__ inL, const bf16* __restrict__ inR,
    const bf16* __restrict__ wihL, const bf16* __restrict__ wihR,
    const float* __restrict__ bihL, const float* __restrict__ bhhL,
    const float* __restrict__ bihR, const float* __restrict__ bhhR,
    float* __restrict__ xgL, float* __restrict__ xgR) {
  const bf16*  inp = blockIdx.z ? inR  : inL;
  const bf16*  wih = blockIdx.z ? wihR : wihL;
  const float* bih = blockIdx.z ? bihR : bihL;
  const float* bhh = blockIdx.z ? bhhR : bhhL;
  float*       xg  = blockIdx.z ? xgR  : xgL;
  int wave = threadIdx.x >> 5, lane = threadIdx.x & 31;
  int m0 = blockIdx.y * 128 + wave * 16;
  int n0 = blockIdx.x * 64;
  constexpr int KS = HID / 32;
  v8f acc[4] = {};
  v16bf a[2], b[2][4];
  a[0] = load_a_frag(inp, HID, m0, 0, lane);
#pragma unroll
  for (int nt = 0; nt < 4; ++nt) b[0][nt] = load_b_frag(wih, HID, n0 + nt * 16, 0, lane);
#pragma unroll 2
  for (int ks = 0; ks < KS; ++ks) {
    const int cur = ks & 1, nxt = cur ^ 1;
    if (ks + 1 < KS) {
      int kb = (ks + 1) * 32;
      __builtin_prefetch(inp + (size_t)(m0 + (lane & 15)) * HID + kb + 32, 0, 1);
      a[nxt] = load_a_frag(inp, HID, m0, kb, lane);
#pragma unroll
      for (int nt = 0; nt < 4; ++nt)
        b[nxt][nt] = load_b_frag(wih, HID, n0 + nt * 16, kb, lane);
    }
#pragma unroll
    for (int nt = 0; nt < 4; ++nt) acc[nt] = wmma_bf16(a[cur], b[cur][nt], acc[nt]);
  }
#pragma unroll
  for (int nt = 0; nt < 4; ++nt) {
    int n = n0 + nt * 16 + (lane & 15);
    float bias = bih[n] + bhh[n];
#pragma unroll
    for (int r = 0; r < 8; ++r) {
      int row = m0 + r + ((lane >> 4) << 3);      // row = b*512 + s
      int b2 = row >> 9, s = row & 511;
      xg[((size_t)s * BB + b2) * G4 + n] = acc[nt][r] + bias;
    }
  }
}

// ---------------------------------------------------------------- LSTM scan
// grid (8 slices, 2 dirs), 256 threads, persistent over 512 steps.
// LDS: gates [64,256] f32 (64KB) | cell [64,64] f32 (16KB) | h stage [64,512] bf16 (64KB)
// h double-buffered in global (bf16); per-step release/acquire atomic barrier.
__global__ __launch_bounds__(256) void lstm_kernel(
    const bf16* __restrict__ whhL, const bf16* __restrict__ whhR,
    const float* __restrict__ xgL, const float* __restrict__ xgR,
    bf16* __restrict__ hbuf,            // [2 dirs][2 bufs][64*512] bf16
    unsigned* __restrict__ counters,    // [2 dirs] (stride 16 dwords)
    float* __restrict__ out) {          // [B,S,1024] f32
  extern __shared__ char smem_raw[];
  float* gates = (float*)smem_raw;          // 64*256
  float* cst   = gates + 64 * 256;          // 64*64
  bf16*  hs    = (bf16*)(cst + 64 * 64);    // 64*512 staged h
  const int slice = blockIdx.x;
  const int dir   = blockIdx.y;
  const bf16*  whh = dir ? whhR : whhL;
  const float* xg  = dir ? xgR  : xgL;
  bf16*     hb  = hbuf + (size_t)dir * 2 * BB * RNNH;
  unsigned* cnt = counters + dir * 16;
  const int hbase = slice * 64;
  const int wave = threadIdx.x >> 5, lane = threadIdx.x & 31;

  for (int i = threadIdx.x; i < 64 * 64; i += 256) cst[i] = 0.0f;
  __syncthreads();

  for (int step = 0; step < SS; ++step) {
    const int t = dir ? (SS - 1 - step) : step;
    const bf16* hin  = hb + (size_t)(step & 1) * BB * RNNH;
    bf16*       hout = hb + (size_t)((step + 1) & 1) * BB * RNNH;
    const float* xgt = xg + (size_t)t * BB * G4;

    // --- stage h [64,512] bf16 into LDS (async) ---------------------------
#pragma unroll
    for (int i = 0; i < 16; ++i) {
      int chunk = threadIdx.x + i * 256;                     // 4096 x 16B = 64KB
      copy16_g2lds(hin + chunk * 8, hs + chunk * 8);
    }
    // prefetch this step's xg slice (one 256B gate-region row per thread)
    {
      int r = threadIdx.x >> 2, g = threadIdx.x & 3;
      __builtin_prefetch(xgt + (size_t)r * G4 + g * RNNH + hbase, 0, 0);
    }
    wait_lds_copies();
    __syncthreads();

    // --- GEMM: [64,512](LDS) x [512, 256 gate cols](global W_hh) -----------
    v8f acc[4][2] = {};
    constexpr int KS = RNNH / 32;
    const int nt0   = wave * 2;
    const int ncol0 = (nt0 >> 2) * RNNH + hbase + (nt0 & 3) * 16;
    const int ncol1 = ((nt0 + 1) >> 2) * RNNH + hbase + ((nt0 + 1) & 3) * 16;
    v16bf bb[2][2];
    bb[0][0] = load_b_frag(whh, RNNH, ncol0, 0, lane);
    bb[0][1] = load_b_frag(whh, RNNH, ncol1, 0, lane);
#pragma unroll 2
    for (int ks = 0; ks < KS; ++ks) {
      const int cur = ks & 1, nxt = cur ^ 1;
      if (ks + 1 < KS) {
        int kb = (ks + 1) * 32;
        bb[nxt][0] = load_b_frag(whh, RNNH, ncol0, kb, lane);
        bb[nxt][1] = load_b_frag(whh, RNNH, ncol1, kb, lane);
      }
      v16bf a[4];
#pragma unroll
      for (int mt = 0; mt < 4; ++mt)
        a[mt] = load_a_frag(hs, RNNH, mt * 16, ks * 32, lane);
#pragma unroll
      for (int j = 0; j < 2; ++j)
#pragma unroll
        for (int mt = 0; mt < 4; ++mt)
          acc[mt][j] = wmma_bf16(a[mt], bb[cur][j], acc[mt][j]);
    }

    // --- gates -> LDS (+ precomputed xg) -----------------------------------
#pragma unroll
    for (int j = 0; j < 2; ++j) {
      int nt   = wave * 2 + j;
      int nloc = nt * 16 + (lane & 15);                      // 0..255
      int nabs = (nt >> 2) * RNNH + hbase + (nt & 3) * 16 + (lane & 15);
#pragma unroll
      for (int mt = 0; mt < 4; ++mt) {
#pragma unroll
        for (int r = 0; r < 8; ++r) {
          int row = mt * 16 + r + ((lane >> 4) << 3);        // batch b
          gates[row * 256 + nloc] = acc[mt][j][r] + xgt[(size_t)row * G4 + nabs];
        }
      }
    }
    __syncthreads();

    // --- elementwise cell update ------------------------------------------
    for (int i = threadIdx.x; i < 64 * 64; i += 256) {
      int row = i >> 6, j = i & 63;
      float ig = gates[row * 256 + j];
      float fg = gates[row * 256 + 64 + j];
      float gg = gates[row * 256 + 128 + j];
      float og = gates[row * 256 + 192 + j];
      float c = sigf(fg) * cst[i] + sigf(ig) * tanhf(gg);
      cst[i] = c;
      float h = sigf(og) * tanhf(c);
      hout[(size_t)row * RNNH + hbase + j] = (bf16)h;
      out[((size_t)row * SS + t) * (2 * RNNH) + dir * RNNH + hbase + j] = h;
    }
    __syncthreads();

    // --- per-step inter-workgroup barrier (8 blocks per direction) --------
    if (threadIdx.x == 0) {
      __threadfence();                                       // release h writes
      atomicAdd(cnt, 1u);
      unsigned target = 8u * (unsigned)(step + 1);
      while (__hip_atomic_load(cnt, __ATOMIC_ACQUIRE, __HIP_MEMORY_SCOPE_AGENT) <
             target)
        __builtin_amdgcn_s_sleep(2);
    }
    __syncthreads();
    __threadfence();                                         // acquire before next read
  }
}

// ---------------------------------------------------------------- launcher
extern "C" void kernel_launch(void* const* d_in, const int* in_sizes, int n_in,
                              void* d_out, int out_size, void* d_ws, size_t ws_size,
                              hipStream_t stream) {
  (void)in_sizes; (void)n_in; (void)out_size; (void)ws_size;
  const int* char_idx  = (const int*)d_in[0];
  const int* ctype_idx = (const int*)d_in[1];
  const int* lbi_idx   = (const int*)d_in[2];
  const int* rbi_idx   = (const int*)d_in[3];
  const int* ext_idx   = (const int*)d_in[4];
  const int* lebi_idx  = (const int*)d_in[5];
  const int* rebi_idx  = (const int*)d_in[6];
  const float* charEmb  = (const float*)d_in[7];
  const float* ctypeEmb = (const float*)d_in[8];
  const float* biEmb    = (const float*)d_in[9];
  const float* extEmb   = (const float*)d_in[10];
  const float* extBiEmb = (const float*)d_in[11];
  const float* W_lin  = (const float*)d_in[12];
  const float* b_lin  = (const float*)d_in[13];
  const float* W_ih_l = (const float*)d_in[14];
  const float* W_hh_l = (const float*)d_in[15];
  const float* b_ih_l = (const float*)d_in[16];
  const float* b_hh_l = (const float*)d_in[17];
  const float* W_ih_r = (const float*)d_in[18];
  const float* W_hh_r = (const float*)d_in[19];
  const float* b_ih_r = (const float*)d_in[20];
  const float* b_hh_r = (const float*)d_in[21];
  float* out = (float*)d_out;

  uint8_t* ws = (uint8_t*)d_ws;
  size_t o = 0;
  auto carve = [&](size_t bytes) -> uint8_t* {
    uint8_t* p = ws + o;
    o = (o + bytes + 255) & ~(size_t)255;
    return p;
  };
  bf16*  catL = (bf16*)carve((size_t)TOK * KPAD * 2);
  bf16*  catR = (bf16*)carve((size_t)TOK * KPAD * 2);
  bf16*  wlin = (bf16*)carve((size_t)HID * KPAD * 2);
  bf16*  wihL = (bf16*)carve((size_t)G4 * HID * 2);
  bf16*  wihR = (bf16*)carve((size_t)G4 * HID * 2);
  bf16*  whhL = (bf16*)carve((size_t)G4 * RNNH * 2);
  bf16*  whhR = (bf16*)carve((size_t)G4 * RNNH * 2);
  bf16*  inL  = (bf16*)carve((size_t)TOK * HID * 2);
  bf16*  inR  = (bf16*)carve((size_t)TOK * HID * 2);
  float* xgL  = (float*)carve((size_t)SS * BB * G4 * 4);
  float* xgR  = (float*)carve((size_t)SS * BB * G4 * 4);
  bf16*  hbuf = (bf16*)carve((size_t)2 * 2 * BB * RNNH * 2);
  unsigned* counters = (unsigned*)carve(128);

  {
    int n = HID * KPAD;
    cvt_pad_wlin_kernel<<<(n + 255) / 256, 256, 0, stream>>>(W_lin, wlin);
    int nw = G4 * HID;
    cvt_kernel<<<(nw + 255) / 256, 256, 0, stream>>>(W_ih_l, wihL, nw);
    cvt_kernel<<<(nw + 255) / 256, 256, 0, stream>>>(W_ih_r, wihR, nw);
    cvt_kernel<<<(nw + 255) / 256, 256, 0, stream>>>(W_hh_l, whhL, nw);
    cvt_kernel<<<(nw + 255) / 256, 256, 0, stream>>>(W_hh_r, whhR, nw);
  }
  {
    int nh = (int)((size_t)2 * 2 * BB * RNNH * 2 / 4);
    zero32_kernel<<<(nh + 255) / 256, 256, 0, stream>>>((uint32_t*)hbuf, nh);
    zero32_kernel<<<1, 64, 0, stream>>>((uint32_t*)counters, 32);
  }
  gather_kernel<<<TOK, 256, 0, stream>>>(char_idx, ctype_idx, lbi_idx, rbi_idx,
                                         ext_idx, lebi_idx, rebi_idx,
                                         charEmb, ctypeEmb, biEmb, extEmb, extBiEmb,
                                         catL, catR);
  {
    dim3 grid(HID / 64, TOK / 128, 2);
    gemm_lin_kernel<<<grid, 256, 0, stream>>>(catL, catR, wlin, b_lin, inL, inR);
  }
  {
    dim3 grid(G4 / 64, TOK / 128, 2);
    gemm_ih_kernel<<<grid, 256, 0, stream>>>(inL, inR, wihL, wihR,
                                             b_ih_l, b_hh_l, b_ih_r, b_hh_r,
                                             xgL, xgR);
  }
  {
    dim3 grid(8, 2);
    size_t shmem = (size_t)(64 * 256 + 64 * 64) * sizeof(float) +
                   (size_t)64 * 512 * sizeof(bf16);   // 144 KB
    lstm_kernel<<<grid, 256, shmem, stream>>>(whhL, whhR, xgL, xgR,
                                              hbuf, counters, out);
  }
}
